// Recommender_33724083208195
// MI455X (gfx1250) — matrix operand order repaired and checked by reading, not compiled, not run on hardware
//
#include <hip/hip_runtime.h>
#include <cmath>

// ---------------------------------------------------------------------------
// Problem constants (from reference)
// ---------------------------------------------------------------------------
#define N_MASHUP 6000
#define N_API    8000
#define N_NODES  14000
#define EMB      128
#define LOC      64
#define NEDGE    448000
#define TAU      0.6f

typedef __attribute__((ext_vector_type(16))) _Float16 v16h;
typedef __attribute__((ext_vector_type(8)))  _Float16 v8h;
typedef __attribute__((ext_vector_type(8)))  float    v8f;

// ---------------------------------------------------------------------------
// Helpers
// ---------------------------------------------------------------------------
__device__ __forceinline__ void atomAddF(float* p, float v) {
    unsafeAtomicAdd(p, v);   // -> global_atomic_add_f32
}

// A-matrix fragment (16x32 f16): lane L holds row M=L&15.
// lanes 0-15: K={k0..k0+15} with k0 = kk
// lanes 16-31: same but k0 = kk+16. Caller passes k0 already adjusted.
__device__ __forceinline__ v16h load_a_frag(const _Float16* row, int k0) {
    v8h lo = *(const v8h*)(row + k0);
    v8h hi = *(const v8h*)(row + k0 + 8);
    v16h r;
#pragma unroll
    for (int i = 0; i < 8; ++i) { r[i] = lo[i]; r[i + 8] = hi[i]; }
    return r;
}

__device__ __forceinline__ v8f wmma_f16(v16h a, v16h b, v8f c) {
    return __builtin_amdgcn_wmma_f32_16x16x32_f16(false, a, false, b,
                                                  (short)0, c, false, false);
}

// ---------------------------------------------------------------------------
// Elementwise / utility kernels
// ---------------------------------------------------------------------------
__global__ void fill_f32(float* p, int n, float v) {
    int i = blockIdx.x * 256 + threadIdx.x;
    if (i < n) p[i] = v;
}

__global__ void degree_kernel(const int* __restrict__ dst, int e, float* __restrict__ deg) {
    int i = blockIdx.x * 256 + threadIdx.x;
    if (i < e) atomAddF(&deg[dst[i]], 1.0f);
}

// one thread per (edge, 4 features): gather feat[src], atomic-add into agg[dst]
__global__ void scatter_kernel(const int* __restrict__ src, const int* __restrict__ dst,
                               const float* __restrict__ feat, float* __restrict__ agg,
                               int F, int e) {
    int fv = F >> 2;
    int t = blockIdx.x * 256 + threadIdx.x;
    if (t >= e * fv) return;
    int ed = t / fv, c = t - ed * fv;
    int s = src[ed], d = dst[ed];
    float4 v = ((const float4*)feat)[(size_t)s * fv + c];
    float* o = agg + (size_t)d * F + c * 4;
    atomAddF(o + 0, v.x); atomAddF(o + 1, v.y);
    atomAddF(o + 2, v.z); atomAddF(o + 3, v.w);
}

// h_neigh = agg / max(deg,1), converted to f16
__global__ void neigh_to_f16(const float* __restrict__ agg, const float* __restrict__ deg,
                             _Float16* __restrict__ dst, int n, int F) {
    int t = blockIdx.x * 256 + threadIdx.x;
    if (t >= n * F) return;
    int row = t / F;
    dst[t] = (_Float16)(agg[t] / fmaxf(deg[row], 1.0f));
}

// strided f32 -> contiguous f16 (rows n, cols d, source leading dim ld)
__global__ void f32_to_f16(const float* __restrict__ src, int ld, int n, int d,
                           _Float16* __restrict__ dst) {
    int t = blockIdx.x * 256 + threadIdx.x;
    if (t >= n * d) return;
    int r = t / d, c = t - r * d;
    dst[t] = (_Float16)src[(size_t)r * ld + c];
}

// W (KxF, f32 row-major) -> Wt (FxK, f16 row-major) so WMMA B-frags are contiguous
__global__ void transpose_f16(const float* __restrict__ W, int K, int F,
                              _Float16* __restrict__ Wt) {
    int t = blockIdx.x * 256 + threadIdx.x;
    if (t >= K * F) return;
    int k = t / F, f = t - k * F;
    Wt[(size_t)f * K + k] = (_Float16)W[t];
}

// row-wise L2 normalize (clamped at 1e-12) -> f16; one wave per row
__global__ __launch_bounds__(128) void norm_rows(const float* __restrict__ src, int ld,
                                                 int n, int d, _Float16* __restrict__ dst) {
    int wave = threadIdx.x >> 5, lane = threadIdx.x & 31;
    int row = blockIdx.x * 4 + wave;
    if (row >= n) return;
    float vals[4]; int cnt = d >> 5;
    float ss = 0.f;
    for (int i = 0; i < cnt; ++i) {
        float v = src[(size_t)row * ld + lane + i * 32];
        vals[i] = v; ss += v * v;
    }
    ss += __shfl_xor(ss, 16); ss += __shfl_xor(ss, 8); ss += __shfl_xor(ss, 4);
    ss += __shfl_xor(ss, 2);  ss += __shfl_xor(ss, 1);
    float inv = 1.0f / fmaxf(sqrtf(ss), 1e-12f);
    for (int i = 0; i < cnt; ++i)
        dst[(size_t)row * d + lane + i * 32] = (_Float16)(vals[i] * inv);
}

// dab[i] = dot(A[i], B[i]); one wave per row
__global__ __launch_bounds__(128) void row_dot(const _Float16* __restrict__ A,
                                               const _Float16* __restrict__ B,
                                               int n, int d, float* __restrict__ out) {
    int wave = threadIdx.x >> 5, lane = threadIdx.x & 31;
    int row = blockIdx.x * 4 + wave;
    if (row >= n) return;
    float s = 0.f;
    for (int c = lane; c < d; c += 32)
        s += (float)A[(size_t)row * d + c] * (float)B[(size_t)row * d + c];
    s += __shfl_xor(s, 16); s += __shfl_xor(s, 8); s += __shfl_xor(s, 4);
    s += __shfl_xor(s, 2);  s += __shfl_xor(s, 1);
    if (lane == 0) out[row] = s;
}

// ---------------------------------------------------------------------------
// WMMA GEMM:  C(f32, ldC) = A1@B1t (+ A2@B2t) + bias, optional relu.
// A*: MxK f16 row-major.  B*t: FxK f16 row-major (pre-transposed weights).
// grid: (ceil(M/64), F/16), block 128 (4 waves; wave owns a 16-row stripe).
// ---------------------------------------------------------------------------
__global__ __launch_bounds__(128) void gemm_wmma(
    const _Float16* __restrict__ A1, const _Float16* __restrict__ B1t,
    const _Float16* __restrict__ A2, const _Float16* __restrict__ B2t,
    const float* __restrict__ bias, float* __restrict__ C,
    int M, int K, int Fout, int ldC, int relu) {
    int wave = threadIdx.x >> 5, lane = threadIdx.x & 31;
    int l15 = lane & 15, hi = lane >> 4;
    int rowbase = blockIdx.x * 64 + wave * 16;
    if (rowbase >= M) return;                       // wave-uniform
    int arow = rowbase + l15; if (arow >= M) arow = M - 1;
    int n = blockIdx.y * 16 + l15;

    const _Float16* ap = A1 + (size_t)arow * K;
    const _Float16* bp = B1t + (size_t)n * K;
    v8f c = {};
    for (int kk = 0; kk < K; kk += 32) {
        v16h a = load_a_frag(ap, kk + (hi ? 16 : 0));
        v16h b = *(const v16h*)(bp + kk + (hi ? 16 : 0));
        c = wmma_f16(a, b, c);
    }
    if (A2) {
        const _Float16* ap2 = A2 + (size_t)arow * K;
        const _Float16* bp2 = B2t + (size_t)n * K;
        for (int kk = 0; kk < K; kk += 32) {
            v16h a = load_a_frag(ap2, kk + (hi ? 16 : 0));
            v16h b = *(const v16h*)(bp2 + kk + (hi ? 16 : 0));
            c = wmma_f16(a, b, c);
        }
    }
    float bv = bias ? bias[n] : 0.f;
#pragma unroll
    for (int r = 0; r < 8; ++r) {                   // c[r] = S[r+8*hi][l15]
        int m = rowbase + r + hi * 8;
        if (m < M) {
            float v = c[r] + bv;
            if (relu) v = fmaxf(v, 0.f);
            C[(size_t)m * ldC + n] = v;
        }
    }
}

// ---------------------------------------------------------------------------
// Fused NCE row-sum:  out0[i] = sum_j exp(Q[i].T0[j]/tau),  out1 likewise vs T1.
// Flash-style: Gram matrix never materialized. One wave per 16 query rows.
// Two independent column tiles in flight -> two independent WMMA accumulation
// chains (XDL ILP) and exp/VALU of one pair overlaps loads/WMMA of the next.
// ---------------------------------------------------------------------------
__global__ __launch_bounds__(128) void nce_rowsum(
    const _Float16* __restrict__ Q, int nQ, int K,
    const _Float16* __restrict__ T0, int nT0, float* __restrict__ out0,
    const _Float16* __restrict__ T1, int nT1, float* __restrict__ out1,
    float invTau) {
    int wave = threadIdx.x >> 5, lane = threadIdx.x & 31;
    int l15 = lane & 15, hi = lane >> 4;
    int rowbase = blockIdx.x * 64 + wave * 16;
    if (rowbase >= nQ) return;                      // wave-uniform
    int qrow = rowbase + l15; if (qrow >= nQ) qrow = nQ - 1;
    const _Float16* qp = Q + (size_t)qrow * K;

    int ksteps = K >> 5;                            // 2 or 4
    v16h aF[4];
    for (int s = 0; s < ksteps; ++s) aF[s] = load_a_frag(qp, s * 32 + hi * 16);

    for (int t = 0; t < 2; ++t) {
        const _Float16* T = t ? T1 : T0;
        int nT = t ? nT1 : nT0;
        float* out = t ? out1 : out0;
        float acc[8] = {0, 0, 0, 0, 0, 0, 0, 0};
        int ntiles = (nT + 15) >> 4;

        int j = 0;
        for (; j + 1 < ntiles; j += 2) {            // two tiles in flight
            int col0 = j * 16 + l15;
            int col1 = col0 + 16;
            int br0 = (col0 < nT) ? col0 : nT - 1;
            int br1 = (col1 < nT) ? col1 : nT - 1;
            const _Float16* bp0 = T + (size_t)br0 * K;
            const _Float16* bp1 = T + (size_t)br1 * K;
            if (j + 2 < ntiles) {                   // warm WGP$/L2 two tiles ahead
                int pc = col0 + 32; if (pc >= nT) pc = nT - 1;
                __builtin_prefetch(T + (size_t)pc * K, 0, 3);
            }
            v8f c0 = {}, c1 = {};
            for (int s = 0; s < ksteps; ++s) {
                v16h b0 = *(const v16h*)(bp0 + s * 32 + hi * 16);
                v16h b1 = *(const v16h*)(bp1 + s * 32 + hi * 16);
                c0 = wmma_f16(aF[s], b0, c0);
                c1 = wmma_f16(aF[s], b1, c1);
            }
            float m0 = (col0 < nT) ? 1.f : 0.f;
            float m1 = (col1 < nT) ? 1.f : 0.f;
#pragma unroll
            for (int r = 0; r < 8; ++r)
                acc[r] += m0 * __expf(c0[r] * invTau) + m1 * __expf(c1[r] * invTau);
        }
        if (j < ntiles) {                           // tail tile
            int col = j * 16 + l15;
            int brow = (col < nT) ? col : nT - 1;
            const _Float16* bp = T + (size_t)brow * K;
            v8f c = {};
            for (int s = 0; s < ksteps; ++s) {
                v16h b = *(const v16h*)(bp + s * 32 + hi * 16);
                c = wmma_f16(aF[s], b, c);
            }
            float mask = (col < nT) ? 1.f : 0.f;
#pragma unroll
            for (int r = 0; r < 8; ++r)
                acc[r] += mask * __expf(c[r] * invTau);
        }
#pragma unroll
        for (int r = 0; r < 8; ++r) {               // reduce across 16 columns
            float v = acc[r];
            v += __shfl_xor(v, 1); v += __shfl_xor(v, 2);
            v += __shfl_xor(v, 4); v += __shfl_xor(v, 8);
            acc[r] = v;
        }
        if (l15 == 0) {
#pragma unroll
            for (int r = 0; r < 8; ++r) {
                int m = rowbase + hi * 8 + r;
                if (m < nQ) out[m] = acc[r];
            }
        }
    }
}

// per-row NCE loss; wave-level reduce first, then 1 atomic per wave (32x fewer)
__global__ void nce_loss(const float* __restrict__ sQQ, const float* __restrict__ sQT,
                         const float* __restrict__ sTT, const float* __restrict__ sTQ,
                         const float* __restrict__ dab, int n, float invTau,
                         float eSelf, float scale, float* __restrict__ acc) {
    int i = blockIdx.x * 256 + threadIdx.x;
    float v = 0.f;
    if (i < n) {
        float d  = dab[i] * invTau;
        float l1 = -d + __logf(fmaxf(sQQ[i] + sQT[i] - eSelf, 1e-30f));
        float l2 = -d + __logf(fmaxf(sTT[i] + sTQ[i] - eSelf, 1e-30f));
        v = scale * (l1 + l2);
    }
    v += __shfl_xor(v, 16); v += __shfl_xor(v, 8); v += __shfl_xor(v, 4);
    v += __shfl_xor(v, 2);  v += __shfl_xor(v, 1);
    if ((threadIdx.x & 31) == 0) atomAddF(acc, v);
}

__global__ void finalize_loss(const float* __restrict__ acc, float* __restrict__ out) {
    out[0] = 0.4f * (acc[0] + acc[1]) + 0.6f * (acc[2] + acc[3]);
}

// ---------------------------------------------------------------------------
// Host orchestration
// ---------------------------------------------------------------------------
static inline dim3 g1(int n) { return dim3((n + 255) / 256); }

extern "C" void kernel_launch(void* const* d_in, const int* in_sizes, int n_in,
                              void* d_out, int out_size, void* d_ws, size_t ws_size,
                              hipStream_t stream) {
    (void)in_sizes; (void)n_in; (void)out_size; (void)ws_size;
    const dim3 B256(256), B128(128);

    // ---- inputs (setup_inputs dict order, params flattened in insertion order)
    const int*   srcE[3] = {(const int*)d_in[0], (const int*)d_in[2], (const int*)d_in[4]};
    const int*   dstE[3] = {(const int*)d_in[1], (const int*)d_in[3], (const int*)d_in[5]};
    const float* x       = (const float*)d_in[6];          // all_embed 14000x128
    // inv:7..12  tag:13..18  sem:19..24  fc1:25..28 fc2:29..32 fc3:33..36 fc4:37..40

    // ---- workspace bump allocator
    char* ws = (char*)d_ws; size_t off = 0;
    auto allocB = [&](size_t bytes) -> char* {
        char* p = ws + off; off = (off + bytes + 255) & ~(size_t)255; return p;
    };
    float*     deg   = (float*)allocB((size_t)N_NODES * 4);
    float*     agg   = (float*)allocB((size_t)N_NODES * EMB * 4);
    float*     hbuf  = (float*)allocB((size_t)N_NODES * EMB * 4);
    float*     mlpH  = (float*)allocB((size_t)N_API * EMB * 4);
    float*     mlpO  = (float*)allocB((size_t)N_API * EMB * 4);
    float*     sQQ   = (float*)allocB((size_t)N_API * 4);
    float*     sQT   = (float*)allocB((size_t)N_API * 4);
    float*     sTT   = (float*)allocB((size_t)N_API * 4);
    float*     sTQ   = (float*)allocB((size_t)N_API * 4);
    float*     dab   = (float*)allocB((size_t)N_API * 4);
    float*     lacc  = (float*)allocB(4 * 4);
    _Float16*  x16   = (_Float16*)allocB((size_t)N_NODES * EMB * 2);
    _Float16*  hn16  = (_Float16*)allocB((size_t)N_NODES * EMB * 2);
    _Float16*  h16   = (_Float16*)allocB((size_t)N_NODES * EMB * 2);
    _Float16*  An16  = (_Float16*)allocB((size_t)N_API * EMB * 2);
    _Float16*  Bn16  = (_Float16*)allocB((size_t)N_API * EMB * 2);
    _Float16*  wt    = (_Float16*)allocB((size_t)262144 * 2);
    size_t wo = 0;
    auto aw = [&](int nh) -> _Float16* { _Float16* p = wt + wo; wo += (size_t)nh; return p; };

    float* outF = (float*)d_out;                    // 14000x256 embeddings + loss
    const float invTau = 1.0f / TAU;
    const float eSelf  = expf(invTau);

    hipLaunchKernelGGL(fill_f32, g1(4), B256, 0, stream, lacc, 4, 0.f);

    // x -> f16 once (shared by all three graphs' layer 1)
    hipLaunchKernelGGL(f32_to_f16, g1(N_NODES * EMB), B256, 0, stream,
                       x, EMB, N_NODES, EMB, x16);

    // ---- 3x two-layer GraphSAGE, layer-2 writes straight into d_out slices
    struct G { int base; int d2; int col; };
    const G gs[3] = {{7, EMB, 128}, {13, LOC, 0}, {19, LOC, 64}};
    for (int gi = 0; gi < 3; ++gi) {
        const G& g = gs[gi];
        const float* c1Ws = (const float*)d_in[g.base + 0];
        const float* c1Wn = (const float*)d_in[g.base + 1];
        const float* c1b  = (const float*)d_in[g.base + 2];
        const float* c2Ws = (const float*)d_in[g.base + 3];
        const float* c2Wn = (const float*)d_in[g.base + 4];
        const float* c2b  = (const float*)d_in[g.base + 5];
        _Float16* c1Wst = aw(EMB * g.d2);  _Float16* c1Wnt = aw(EMB * g.d2);
        _Float16* c2Wst = aw(g.d2 * g.d2); _Float16* c2Wnt = aw(g.d2 * g.d2);
        hipLaunchKernelGGL(transpose_f16, g1(EMB * g.d2),  B256, 0, stream, c1Ws, EMB,  g.d2, c1Wst);
        hipLaunchKernelGGL(transpose_f16, g1(EMB * g.d2),  B256, 0, stream, c1Wn, EMB,  g.d2, c1Wnt);
        hipLaunchKernelGGL(transpose_f16, g1(g.d2 * g.d2), B256, 0, stream, c2Ws, g.d2, g.d2, c2Wst);
        hipLaunchKernelGGL(transpose_f16, g1(g.d2 * g.d2), B256, 0, stream, c2Wn, g.d2, g.d2, c2Wnt);

        hipLaunchKernelGGL(fill_f32, g1(N_NODES), B256, 0, stream, deg, N_NODES, 0.f);
        hipLaunchKernelGGL(degree_kernel, g1(NEDGE), B256, 0, stream, dstE[gi], NEDGE, deg);

        // layer 1: F = 128
        hipLaunchKernelGGL(fill_f32, g1(N_NODES * EMB), B256, 0, stream, agg, N_NODES * EMB, 0.f);
        hipLaunchKernelGGL(scatter_kernel, g1(NEDGE * (EMB / 4)), B256, 0, stream,
                           srcE[gi], dstE[gi], x, agg, EMB, NEDGE);
        hipLaunchKernelGGL(neigh_to_f16, g1(N_NODES * EMB), B256, 0, stream,
                           agg, deg, hn16, N_NODES, EMB);
        hipLaunchKernelGGL(gemm_wmma, dim3((N_NODES + 63) / 64, g.d2 / 16), B128, 0, stream,
                           x16, c1Wst, hn16, c1Wnt, c1b, hbuf,
                           N_NODES, EMB, g.d2, g.d2, 1);

        // layer 2: F = d2
        hipLaunchKernelGGL(f32_to_f16, g1(N_NODES * g.d2), B256, 0, stream,
                           hbuf, g.d2, N_NODES, g.d2, h16);
        hipLaunchKernelGGL(fill_f32, g1(N_NODES * g.d2), B256, 0, stream, agg, N_NODES * g.d2, 0.f);
        hipLaunchKernelGGL(scatter_kernel, g1(NEDGE * (g.d2 / 4)), B256, 0, stream,
                           srcE[gi], dstE[gi], hbuf, agg, g.d2, NEDGE);
        hipLaunchKernelGGL(neigh_to_f16, g1(N_NODES * g.d2), B256, 0, stream,
                           agg, deg, hn16, N_NODES, g.d2);
        hipLaunchKernelGGL(gemm_wmma, dim3((N_NODES + 63) / 64, g.d2 / 16), B128, 0, stream,
                           h16, c2Wst, hn16, c2Wnt, c2b, outF + g.col,
                           N_NODES, g.d2, g.d2, 256, 0);
    }

    // ---- 4 contrastive losses (MLP -> normalize -> fused NCE row-sums)
    struct L { int rowOff, n, d, colA, colB, base, idx; };
    const L ls[4] = {
        {0,        N_MASHUP, LOC, 0, 64,  25, 0},   // fc1: tag_m vs sem_m
        {N_MASHUP, N_API,    LOC, 0, 64,  29, 1},   // fc2: tag_a vs sem_a
        {0,        N_MASHUP, EMB, 0, 128, 33, 2},   // fc3: [tag_m|sem_m] vs inv_m
        {N_MASHUP, N_API,    EMB, 0, 128, 37, 3},   // fc4: [tag_a|sem_a] vs inv_a
    };
    for (int li = 0; li < 4; ++li) {
        const L& c = ls[li];
        const float* W1 = (const float*)d_in[c.base + 0];
        const float* b1 = (const float*)d_in[c.base + 1];
        const float* W2 = (const float*)d_in[c.base + 2];
        const float* b2 = (const float*)d_in[c.base + 3];
        _Float16* W1t = aw(c.d * c.d); _Float16* W2t = aw(c.d * c.d);
        hipLaunchKernelGGL(transpose_f16, g1(c.d * c.d), B256, 0, stream, W1, c.d, c.d, W1t);
        hipLaunchKernelGGL(transpose_f16, g1(c.d * c.d), B256, 0, stream, W2, c.d, c.d, W2t);

        for (int side = 0; side < 2; ++side) {
            const float* srcp = outF + (size_t)c.rowOff * 256 + (side ? c.colB : c.colA);
            _Float16* nout = side ? Bn16 : An16;
            hipLaunchKernelGGL(f32_to_f16, g1(c.n * c.d), B256, 0, stream,
                               srcp, 256, c.n, c.d, x16);
            hipLaunchKernelGGL(gemm_wmma, dim3((c.n + 63) / 64, c.d / 16), B128, 0, stream,
                               x16, W1t, (const _Float16*)nullptr, (const _Float16*)nullptr,
                               b1, mlpH, c.n, c.d, c.d, c.d, 1);
            hipLaunchKernelGGL(f32_to_f16, g1(c.n * c.d), B256, 0, stream,
                               mlpH, c.d, c.n, c.d, h16);
            hipLaunchKernelGGL(gemm_wmma, dim3((c.n + 63) / 64, c.d / 16), B128, 0, stream,
                               h16, W2t, (const _Float16*)nullptr, (const _Float16*)nullptr,
                               b2, mlpO, c.n, c.d, c.d, c.d, 0);
            hipLaunchKernelGGL(norm_rows, dim3((c.n + 3) / 4), B128, 0, stream,
                               mlpO, c.d, c.n, c.d, nout);
        }
        hipLaunchKernelGGL(row_dot, dim3((c.n + 3) / 4), B128, 0, stream,
                           An16, Bn16, c.n, c.d, dab);
        hipLaunchKernelGGL(nce_rowsum, dim3((c.n + 63) / 64), B128, 0, stream,
                           An16, c.n, c.d, An16, c.n, sQQ, Bn16, c.n, sQT, invTau);
        hipLaunchKernelGGL(nce_rowsum, dim3((c.n + 63) / 64), B128, 0, stream,
                           Bn16, c.n, c.d, Bn16, c.n, sTT, An16, c.n, sTQ, invTau);
        hipLaunchKernelGGL(nce_loss, g1(c.n), B256, 0, stream,
                           sQQ, sQT, sTT, sTQ, dab, c.n, invTau, eSelf,
                           0.5f / (float)c.n, lacc + c.idx);
    }

    hipLaunchKernelGGL(finalize_loss, dim3(1), dim3(1), 0, stream,
                       lacc, outF + (size_t)N_NODES * 256);
}
